// GAEAttn_61323543052778
// MI455X (gfx1250) — compile-verified, matrix operand order checked by reading, not run on hardware
//
#include <hip/hip_runtime.h>
#include <hip/hip_bf16.h>

// ---------------------------------------------------------------------------
// GAT (3 layers, 2 heads) + pairwise decoder for MI455X (gfx1250, wave32).
// GEMMs run on the WMMA pipe (v_wmma_f32_16x16x32_bf16, f32 accumulate) with
// one wave owning a 16x176 output strip (11 WMMA tiles, A-fragment reused).
// All 11 B fragments of a K-step are preloaded into distinct registers so the
// load clause overlaps the WMMA stream (partial s_wait_loadcnt, no full waits).
// Edge gather/scatter and decoder are bandwidth-shaped wave-per-item kernels.
// ---------------------------------------------------------------------------

typedef __bf16 bf16_t;
typedef __attribute__((ext_vector_type(16))) __bf16 v16bf;
typedef __attribute__((ext_vector_type(8)))  __bf16 v8bf;
typedef __attribute__((ext_vector_type(8)))  float  v8f;

#define GAT_N   20000
#define GAT_F   256
#define GAT_E   160000
#define GAT_M   100000
#define GAT_H   2
#define GAT_U   350
#define GAT_UP  352          // padded UNITS (22 tiles of 16; 11*32)
#define GAT_CP  (2*GAT_UP)   // padded concat width = 704 (22 k-steps of 32)
#define NT_WAVE 11           // N-tiles per wave (2 waves cover all 22)

__device__ __forceinline__ float waveReduceSum(float v) {
    #pragma unroll
    for (int off = 16; off > 0; off >>= 1) v += __shfl_xor(v, off, 32);
    return v;
}

// ---- Weight repack: K[h][kreal][350] (f32) -> WT[h][n=352][kpad] (bf16) ----
// kpad rows grouped in blocks of 352; block b, slot j -> original row b*350+j
// when j<350, else zero.  Columns n>=350 are zero (pads stay inert end-to-end).
__global__ void pad_w_kernel(const float* __restrict__ K, bf16_t* __restrict__ WT,
                             int Kreal, int Kpad) {
    long long i = (long long)blockIdx.x * blockDim.x + threadIdx.x;
    long long tot = (long long)GAT_H * GAT_UP * Kpad;
    if (i >= tot) return;
    int kp = (int)(i % Kpad);
    int n  = (int)((i / Kpad) % GAT_UP);
    int h  = (int)(i / ((long long)Kpad * GAT_UP));
    int blk = kp / GAT_UP;
    int j   = kp - blk * GAT_UP;
    int ko  = blk * GAT_U + j;
    float v = 0.0f;
    if (j < GAT_U && ko < Kreal && n < GAT_U)
        v = K[((long long)h * Kreal + ko) * GAT_U + n];
    WT[i] = (bf16_t)v;
}

// ---- Attention vector pad: A[h][2U][1] -> asrc[h][352], adst[h][352] ----
__global__ void pad_a_kernel(const float* __restrict__ A,
                             float* __restrict__ asrc, float* __restrict__ adst) {
    int i = blockIdx.x * blockDim.x + threadIdx.x;
    if (i >= GAT_H * GAT_UP) return;
    int u = i % GAT_UP;
    int h = i / GAT_UP;
    asrc[i] = (u < GAT_U) ? A[(long long)h * (2 * GAT_U) + u]          : 0.0f;
    adst[i] = (u < GAT_U) ? A[(long long)h * (2 * GAT_U) + GAT_U + u]  : 0.0f;
}

// ---- WMMA GEMM: H[head][m][u] = X[m][:] @ W[head][:][u] ----
// Wave strip-mines 16 rows x 176 cols.  Per K-step:
//   1) 4x b128 A loads (f32) issued first,
//   2) 22x b128 B loads (bf16) into 11 distinct fragment buffers,
//   3) A cvt (waits only on A: loads complete in issue order),
//   4) 11 independent v_wmma_f32_16x16x32_bf16 draining the B clause with
//      partial loadcnt waits.
template <int KP>
__global__ void gemm_wmma_kernel(const float* __restrict__ X, int ldx,
                                 const bf16_t* __restrict__ WT,
                                 float* __restrict__ Hout) {
    const int lane  = threadIdx.x & 31;
    const int wave  = threadIdx.x >> 5;               // 0..7
    const int mTile = blockIdx.x * 4 + (wave >> 1);
    const int nBase = (wave & 1) * NT_WAVE;           // 0 or 11
    const int head  = blockIdx.z;
    if (mTile >= GAT_N / 16) return;                  // wave-uniform: EXEC stays all-1s

    const int half = lane >> 4;                       // K-base 0 / 8 within fragment
    const int l15  = lane & 15;
    const float*  xrow  = X  + (long long)(mTile * 16 + l15) * ldx + half * 8;
    const bf16_t* wbase = WT + ((long long)head * GAT_UP + l15) * KP + half * 8
                             + (long long)nBase * 16 * KP;

    v8f acc[NT_WAVE] = {};
    for (int k0 = 0; k0 < KP; k0 += 32) {
        // (1) A loads: two contiguous 8-elem K runs -> 4x b128.
        const float4 x0 = *(const float4*)(xrow + k0);
        const float4 x1 = *(const float4*)(xrow + k0 + 4);
        const float4 x2 = *(const float4*)(xrow + k0 + 16);
        const float4 x3 = *(const float4*)(xrow + k0 + 20);
        // (2) B loads: all 11 fragments, distinct buffers, 2x b128 each.
        v8bf b0s[NT_WAVE], b1s[NT_WAVE];
        #pragma unroll
        for (int t = 0; t < NT_WAVE; ++t) {
            const bf16_t* wrow = wbase + (long long)t * 16 * KP;
            b0s[t] = *(const v8bf*)(wrow + k0);
            b1s[t] = *(const v8bf*)(wrow + k0 + 16);
        }
        // (3) A f32->bf16 conversion (depends only on the 4 A loads).
        v16bf a;
        a[0]  = (bf16_t)x0.x; a[1]  = (bf16_t)x0.y; a[2]  = (bf16_t)x0.z; a[3]  = (bf16_t)x0.w;
        a[4]  = (bf16_t)x1.x; a[5]  = (bf16_t)x1.y; a[6]  = (bf16_t)x1.z; a[7]  = (bf16_t)x1.w;
        a[8]  = (bf16_t)x2.x; a[9]  = (bf16_t)x2.y; a[10] = (bf16_t)x2.z; a[11] = (bf16_t)x2.w;
        a[12] = (bf16_t)x3.x; a[13] = (bf16_t)x3.y; a[14] = (bf16_t)x3.z; a[15] = (bf16_t)x3.w;
        // (4) WMMA stream: each tile waits only for its own 2 B loads.
        #pragma unroll
        for (int t = 0; t < NT_WAVE; ++t) {
            v16bf b;
            #pragma unroll
            for (int q = 0; q < 8; ++q) { b[q] = b0s[t][q]; b[8 + q] = b1s[t][q]; }
            acc[t] = __builtin_amdgcn_wmma_f32_16x16x32_bf16(
                         false, a, false, b, (short)0, acc[t], false, false);
        }
    }
    // C/D layout: VGPR r, lanes 0-15 -> M=r, lanes 16-31 -> M=8+r; N=lane&15.
    float* hbase = Hout + ((long long)head * GAT_N + mTile * 16 + half * 8) * GAT_UP + l15;
    #pragma unroll
    for (int t = 0; t < NT_WAVE; ++t) {
        float* hp = hbase + (nBase + t) * 16;
        #pragma unroll
        for (int r = 0; r < 8; ++r)
            hp[(long long)r * GAT_UP] = acc[t][r];
    }
}

// ---- Per-node attention scores: s = h . a  (wave per (head,node)) ----
__global__ void scores_kernel(const float* __restrict__ H,
                              const float* __restrict__ asrc,
                              const float* __restrict__ adst,
                              float* __restrict__ s_src, float* __restrict__ s_dst) {
    const int lane = threadIdx.x & 31;
    const int node = blockIdx.x * 8 + (threadIdx.x >> 5);
    const int head = blockIdx.y;
    if (node >= GAT_N) return;
    const float* hrow = H + ((long long)head * GAT_N + node) * GAT_UP;
    float ps = 0.0f, pd = 0.0f;
    #pragma unroll
    for (int u = lane; u < GAT_UP; u += 32) {
        const float hv = hrow[u];
        ps += hv * asrc[head * GAT_UP + u];
        pd += hv * adst[head * GAT_UP + u];
    }
    ps = waveReduceSum(ps);
    pd = waveReduceSum(pd);
    if (lane == 0) {
        s_src[head * GAT_N + node] = ps;
        s_dst[head * GAT_N + node] = pd;
    }
}

// ---- Edge pass 1: leaky-relu -> clip -> exp, accumulate softmax denominator --
__global__ void edge_score_kernel(const int* __restrict__ edges,
                                  const float* __restrict__ s_src,
                                  const float* __restrict__ s_dst,
                                  float* __restrict__ esc, float* __restrict__ denom) {
    const int e = blockIdx.x * blockDim.x + threadIdx.x;
    const int head = blockIdx.y;
    if (e >= GAT_E) return;
    const int src = edges[2 * e];
    const int dst = edges[2 * e + 1];
    float sc = s_src[head * GAT_N + src] + s_dst[head * GAT_N + dst];
    sc = (sc > 0.0f) ? sc : 0.2f * sc;           // leaky_relu(0.2)
    sc = fminf(fmaxf(sc, -2.0f), 2.0f);          // clip
    sc = __expf(sc);
    esc[(long long)head * GAT_E + e] = sc;
    atomicAdd(&denom[head * GAT_N + src], sc);
}

// ---- Edge pass 2: weighted scatter-add of h[dst] into act[src] (wave/edge) --
__global__ void edge_agg_kernel(const int* __restrict__ edges,
                                const float* __restrict__ esc,
                                const float* __restrict__ denom,
                                const float* __restrict__ H,
                                float* __restrict__ actOut) {
    const int lane = threadIdx.x & 31;
    const int e    = blockIdx.x * 8 + (threadIdx.x >> 5);
    const int head = blockIdx.y;
    if (e >= GAT_E) return;
    const int src = edges[2 * e];
    const int dst = edges[2 * e + 1];
    const float w = esc[(long long)head * GAT_E + e] / denom[head * GAT_N + src];
    const float* hrow = H + ((long long)head * GAT_N + dst) * GAT_UP;
    float* orow = actOut + (long long)src * GAT_CP + head * GAT_UP;
    __builtin_prefetch(hrow, 0, 1);              // global_prefetch_b8
    #pragma unroll
    for (int u = lane; u < GAT_UP; u += 32)      // 352 = 11 * 32, no remainder
        atomicAdd(&orow[u], hrow[u] * w);
}

// ---- ReLU on the concatenated activation buffer ----
__global__ void relu_kernel(float* __restrict__ a, long long n) {
    long long i = (long long)blockIdx.x * blockDim.x + threadIdx.x;
    if (i < n) a[i] = fmaxf(a[i], 0.0f);
}

// ---- Decoder: out[m] = |act[r]-act[c]| . dec_w + dec_b (wave per pair) ----
__global__ void decoder_kernel(const float* __restrict__ act,
                               const int* __restrict__ r_idx,
                               const int* __restrict__ c_idx,
                               const float* __restrict__ dec_w,
                               const float* __restrict__ dec_b,
                               float* __restrict__ out) {
    const int lane = threadIdx.x & 31;
    const int m    = blockIdx.x * 8 + (threadIdx.x >> 5);
    if (m >= GAT_M) return;
    const float* ra = act + (long long)r_idx[m] * GAT_CP;
    const float* ca = act + (long long)c_idx[m] * GAT_CP;
    float p = 0.0f;
    #pragma unroll
    for (int col = lane; col < GAT_CP; col += 32) {   // 704 = 22 * 32
        const int h = (col >= GAT_UP) ? 1 : 0;
        const int j = col - h * GAT_UP;
        if (j < GAT_U)
            p += fabsf(ra[col] - ca[col]) * dec_w[h * GAT_U + j];
    }
    p = waveReduceSum(p);
    if (lane == 0) out[m] = p + dec_b[0];
}

// ---------------------------------------------------------------------------
extern "C" void kernel_launch(void* const* d_in, const int* in_sizes, int n_in,
                              void* d_out, int out_size, void* d_ws, size_t ws_size,
                              hipStream_t stream) {
    const float* node_feats = (const float*)d_in[0];
    const int*   edges      = (const int*)  d_in[1];
    const int*   r_idx      = (const int*)  d_in[2];
    const int*   c_idx      = (const int*)  d_in[3];
    const float* kW[3] = { (const float*)d_in[4], (const float*)d_in[6], (const float*)d_in[8] };
    const float* aW[3] = { (const float*)d_in[5], (const float*)d_in[7], (const float*)d_in[9] };
    const float* dec_w = (const float*)d_in[10];
    const float* dec_b = (const float*)d_in[11];
    float* out = (float*)d_out;

    // ---- workspace carve-up (256B aligned) ----
    char*  base = (char*)d_ws;
    size_t off  = 0;
    auto carve = [&](size_t bytes) { void* p = base + off; off += (bytes + 255) & ~(size_t)255; return p; };
    bf16_t* WT    = (bf16_t*)carve((size_t)GAT_H * GAT_UP * GAT_CP * sizeof(bf16_t)); // max layer
    float*  asrc  = (float*) carve((size_t)GAT_H * GAT_UP * sizeof(float));
    float*  adst  = (float*) carve((size_t)GAT_H * GAT_UP * sizeof(float));
    float*  Hbuf  = (float*) carve((size_t)GAT_H * GAT_N * GAT_UP * sizeof(float));
    float*  act   = (float*) carve((size_t)GAT_N * GAT_CP * sizeof(float));
    float*  s_src = (float*) carve((size_t)GAT_H * GAT_N * sizeof(float));
    float*  s_dst = (float*) carve((size_t)GAT_H * GAT_N * sizeof(float));
    float*  denom = (float*) carve((size_t)GAT_H * GAT_N * sizeof(float));
    float*  esc   = (float*) carve((size_t)GAT_H * GAT_E * sizeof(float));
    (void)ws_size; (void)n_in; (void)in_sizes; (void)out_size;

    const int Kreal[3] = { GAT_F, 2 * GAT_U, 2 * GAT_U };
    const int Kpad [3] = { GAT_F, GAT_CP,    GAT_CP    };
    const unsigned gemmBlocksX = (GAT_N / 16 + 3) / 4;   // 4 mTiles per 8-wave block

    for (int l = 0; l < 3; ++l) {
        // 1) repack weights/attention vectors (padded, bf16 / f32)
        {
            long long tot = (long long)GAT_H * GAT_UP * Kpad[l];
            pad_w_kernel<<<dim3((unsigned)((tot + 255) / 256)), 256, 0, stream>>>(kW[l], WT, Kreal[l], Kpad[l]);
            pad_a_kernel<<<dim3((GAT_H * GAT_UP + 255) / 256), 256, 0, stream>>>(aW[l], asrc, adst);
        }
        // 2) h = X @ W  (WMMA bf16, f32 accumulate)
        if (l == 0) {
            gemm_wmma_kernel<GAT_F><<<dim3(gemmBlocksX, 1, GAT_H), 256, 0, stream>>>(
                node_feats, GAT_F, WT, Hbuf);
        } else {
            gemm_wmma_kernel<GAT_CP><<<dim3(gemmBlocksX, 1, GAT_H), 256, 0, stream>>>(
                act, GAT_CP, WT, Hbuf);
        }
        // 3) attention scores
        scores_kernel<<<dim3(GAT_N / 8, GAT_H), 256, 0, stream>>>(Hbuf, asrc, adst, s_src, s_dst);
        // 4) zero denominators + output activations (act input already consumed)
        hipMemsetAsync(denom, 0, (size_t)GAT_H * GAT_N * sizeof(float), stream);
        hipMemsetAsync(act,   0, (size_t)GAT_N * GAT_CP * sizeof(float), stream);
        // 5) edge softmax numerators + denominator scatter
        edge_score_kernel<<<dim3((GAT_E + 255) / 256, GAT_H), 256, 0, stream>>>(edges, s_src, s_dst, esc, denom);
        // 6) weighted message scatter-add
        edge_agg_kernel<<<dim3(GAT_E / 8, GAT_H), 256, 0, stream>>>(edges, esc, denom, Hbuf, act);
        // 7) relu
        relu_kernel<<<dim3((unsigned)(((long long)GAT_N * GAT_CP + 255) / 256)), 256, 0, stream>>>(
            act, (long long)GAT_N * GAT_CP);
    }

    // 8) decoder over node pairs
    decoder_kernel<<<dim3(GAT_M / 8), 256, 0, stream>>>(act, r_idx, c_idx, dec_w, dec_b, out);
}
